// TransformerEncoder_34033320853825
// MI455X (gfx1250) — compile-verified, matrix operand order checked by reading, not run on hardware
//
#include <hip/hip_runtime.h>
#include <hip/hip_bf16.h>

// ---------------------------------------------------------------------------
// Types for CDNA5 WMMA (wave32)
// ---------------------------------------------------------------------------
typedef __bf16 bf16;
typedef __attribute__((ext_vector_type(16))) __bf16 v16bf;
typedef __attribute__((ext_vector_type(8)))  __bf16 v8bf;
typedef __attribute__((ext_vector_type(8)))  float  v8f;

union FragBF16 { v16bf f; v8bf h[2]; };

__device__ __forceinline__ v8f wmma_bf16(const FragBF16& a, const FragBF16& b, v8f c) {
  // D(f32 16x16) = A(bf16 16x32) * B(bf16 32x16) + C
  return __builtin_amdgcn_wmma_f32_16x16x32_bf16(false, a.f, false, b.f,
                                                 (short)0, c, false, false);
}

// ---------------------------------------------------------------------------
// Tensor Data Mover support (probe-confirmed builtin; arity differs per
// toolchain -- see CDNA5_HIP.md).  Falls back to cooperative loads if absent.
// ---------------------------------------------------------------------------
#if defined(__has_builtin)
#if __has_builtin(__builtin_amdgcn_tensor_load_to_lds) && \
    __has_builtin(__builtin_amdgcn_s_wait_tensorcnt)
#define HAS_TDM 1
#endif
#endif
#ifndef HAS_TDM
#define HAS_TDM 0
#endif

#if HAS_TDM
typedef __attribute__((ext_vector_type(4))) unsigned tdmv4u;
typedef __attribute__((ext_vector_type(8))) int      tdmv8i;
typedef __attribute__((ext_vector_type(4))) int      tdmv4i;

// Issue one TDM 2D tile load: tileW x tileH elements of 2 bytes, row stride
// `strideElems`, into LDS byte offset `ldsByteOff` with LDS row padding
// (padInterval/padAmount are the D# codes: interval dwords = 2<<i, pad
// dwords = code+1).  Wave-uniform; tracked on TENSORcnt.
__device__ __forceinline__ void tdm_load_2d(unsigned ldsByteOff, const void* gptr,
                                            unsigned tileW, unsigned tileH,
                                            unsigned strideElems,
                                            unsigned padInterval, unsigned padAmount) {
  unsigned long long ga = (unsigned long long)gptr;
  tdmv4u g0;
  g0[0] = 1u;                                    // count=1 (valid user descriptor)
  g0[1] = ldsByteOff;                            // lds_addr (bytes)
  g0[2] = (unsigned)ga;                          // global_addr lo
  g0[3] = (unsigned)((ga >> 32) & 0x1FFFFFFu) | (2u << 30);  // ga hi | type=2
  const unsigned td0 = 1u << 24, td1 = 1u << 24; // large tensor dims: no OOB clip
  tdmv8i g1;
  g1[0] = (int)((1u << 16)                       // data_size = 2 bytes
              | (1u << 20)                       // pad_enable
              | (padInterval << 22) | (padAmount << 25));
  g1[1] = (int)((td0 & 0xFFFFu) << 16);          // tensor_dim0 lo16 (atomic addr 0)
  g1[2] = (int)(((td0 >> 16) & 0xFFFFu) | ((td1 & 0xFFFFu) << 16));
  g1[3] = (int)(((td1 >> 16) & 0xFFFFu) | (tileW << 16));  // tile_dim0
  g1[4] = (int)tileH;                            // tile_dim1 (tile_dim2 = 0)
  g1[5] = (int)strideElems;                      // tensor_dim0_stride lo32
  g1[6] = 0;
  g1[7] = 0;
  tdmv4i z = {0, 0, 0, 0};
#if __clang_major__ >= 23
  tdmv8i z8 = {0, 0, 0, 0, 0, 0, 0, 0};
  __builtin_amdgcn_tensor_load_to_lds(g0, g1, z, z, z8, 0);
#else
  __builtin_amdgcn_tensor_load_to_lds(g0, g1, z, z, 0);
#endif
}
#endif  // HAS_TDM

// The TDM writes LDS behind the compiler's back (it only sees an integer LDS
// offset), so loads from TDM-filled LDS would otherwise be folded to undef.
// Escape the LDS object into an asm with a memory clobber ("the asm may have
// written it"), but through a THROWAWAY pointer copy: the real loads keep
// using the original __shared__ array so ISel still knows the address space
// and emits ds_load_b128 (not flat_load).
__device__ __forceinline__ void lds_dirty(const void* p) {
  void* q = (void*)p;
  asm volatile("" : "+v"(q) :: "memory");
}

// Problem constants (from the reference)
#define BB 4
#define SS 1024
#define DD 1024
#define FF 4096
#define VV 32000
#define LL 2
#define HH 16
#define DK 64
#define BS (BB * SS)   // 4096 rows

// ---------------------------------------------------------------------------
// f32 -> bf16 conversion (8 elements / thread)
// ---------------------------------------------------------------------------
__global__ __launch_bounds__(256) void cvt_f32_to_bf16(const float* __restrict__ src,
                                                       bf16* __restrict__ dst, long n) {
  long i = ((long)blockIdx.x * 256 + threadIdx.x) * 8;
  if (i >= n) return;
  float4 a = *(const float4*)&src[i];
  float4 b = *(const float4*)&src[i + 4];
  v8bf o;
  o[0] = (bf16)a.x; o[1] = (bf16)a.y; o[2] = (bf16)a.z; o[3] = (bf16)a.w;
  o[4] = (bf16)b.x; o[5] = (bf16)b.y; o[6] = (bf16)b.z; o[7] = (bf16)b.w;
  *(v8bf*)&dst[i] = o;
}

// ---------------------------------------------------------------------------
// Embedding * sqrt(D) + sinusoidal positional encoding  (one float4 / thread)
// ---------------------------------------------------------------------------
__global__ __launch_bounds__(256) void embed_pos(const int* __restrict__ tokens,
                                                 const float* __restrict__ emb,
                                                 float* __restrict__ x) {
  long i = (long)blockIdx.x * 256 + threadIdx.x;       // 0 .. BS*D/4-1
  long row = i >> 8;                                   // D/4 = 256 chunks per row
  int  d   = (int)(i & 255) * 4;
  int  tok = tokens[row];
  float s  = (float)(row & (SS - 1));
  const float C = 9.210340371976184f / (float)DD;      // ln(10000)/D
  float a0 = s * __expf(-(float)d * C);
  float a1 = s * __expf(-(float)(d + 2) * C);
  float4 e = *(const float4*)&emb[(long)tok * DD + d];
  float4 o;
  o.x = e.x * 32.0f + __sinf(a0);
  o.y = e.y * 32.0f + __cosf(a0);
  o.z = e.z * 32.0f + __sinf(a1);
  o.w = e.w * 32.0f + __cosf(a1);
  *(float4*)&x[row * DD + d] = o;
}

// ---------------------------------------------------------------------------
// LayerNorm (torch-style: unbiased std, /(std+eps)) -> bf16.  One wave per row.
// ---------------------------------------------------------------------------
__global__ __launch_bounds__(256) void layernorm_bf16(const float* __restrict__ x,
                                                      const float* __restrict__ ga,
                                                      const float* __restrict__ gb,
                                                      bf16* __restrict__ y) {
  int wave = threadIdx.x >> 5, lane = threadIdx.x & 31;
  long row = (long)blockIdx.x * 8 + wave;
  const float* xr = x + row * DD;
  float4 v[8];
  float s = 0.f, sq = 0.f;
#pragma unroll
  for (int i = 0; i < 8; ++i) {
    v[i] = *(const float4*)&xr[(i * 32 + lane) * 4];
    s  += v[i].x + v[i].y + v[i].z + v[i].w;
    sq += v[i].x * v[i].x + v[i].y * v[i].y + v[i].z * v[i].z + v[i].w * v[i].w;
  }
#pragma unroll
  for (int m = 1; m < 32; m <<= 1) {
    s  += __shfl_xor(s,  m, 32);
    sq += __shfl_xor(sq, m, 32);
  }
  float mean = s * (1.f / (float)DD);
  float var  = (sq - s * mean) * (1.f / (float)(DD - 1));
  float rs   = 1.f / (sqrtf(fmaxf(var, 0.f)) + 1e-6f);
  bf16* yr = y + row * DD;
#pragma unroll
  for (int i = 0; i < 8; ++i) {
    int col = (i * 32 + lane) * 4;
    float4 a = *(const float4*)&ga[col];
    float4 b = *(const float4*)&gb[col];
    yr[col + 0] = (bf16)(a.x * (v[i].x - mean) * rs + b.x);
    yr[col + 1] = (bf16)(a.y * (v[i].y - mean) * rs + b.y);
    yr[col + 2] = (bf16)(a.z * (v[i].z - mean) * rs + b.z);
    yr[col + 3] = (bf16)(a.w * (v[i].w - mean) * rs + b.w);
  }
}

// ---------------------------------------------------------------------------
// WMMA GEMM:  C[M,N] = A[M,K](bf16) * W[N,K]^T(bf16) (+bias) (+relu) (+resid)
// Block: 256 thr = 8 waves; block tile 128x128; wave tile 32x64; K-step 32.
// TDM path: double-buffered LDS, wave0 issues tensor_load_to_lds for the
// next K tile while all waves run WMMA on the current one.
// ---------------------------------------------------------------------------
#define GT_K 32
#define GLD  40     // padded LDS row stride (bf16): 16 dwords data + 4 dwords pad
#define GEMM_TILE_BYTES (128 * GLD * 2)   // 10240

__global__ __launch_bounds__(256) void gemm_bf16_wmma(
    const bf16* __restrict__ A, const bf16* __restrict__ W,
    const float* __restrict__ bias, const float* resid,
    float* outF, bf16* outB,
    int M, int N, int K, int relu) {
  __shared__ __align__(128) char smem[4 * GEMM_TILE_BYTES];
  // Buffer layout: [A0][B0][A1][B1]; pointers computed by arithmetic only
  // (no LDS-pointer arrays: their constant initializers break the linker).

  int tid = threadIdx.x, lane = tid & 31, wave = tid >> 5;
  int wm = wave & 3, wn = wave >> 2;
  int half = lane >> 4, l16 = lane & 15;
  long rowBase = (long)blockIdx.y * 128;
  long colBase = (long)blockIdx.x * 128;

  v8f acc[2][4] = {};

#if HAS_TDM
  const bf16* Abase = A + rowBase * K;
  const bf16* Wbase = W + colBase * K;
  // Prologue: stage tile 0 into buffer 0.  pad codes: interval 16 dw, pad 4 dw.
  if (wave == 0) {
    tdm_load_2d(0u,                        Abase, GT_K, 128, K, 3u, 3u);
    tdm_load_2d((unsigned)GEMM_TILE_BYTES, Wbase, GT_K, 128, K, 3u, 3u);
  }
  for (int k0 = 0; k0 < K; k0 += GT_K) {
    unsigned cur = (unsigned)(k0 >> 5) & 1u;
    if (wave == 0) {
      if (k0 + GT_K < K) {          // issue next pair into the other buffer
        unsigned nb = (cur ^ 1u) * (unsigned)(2 * GEMM_TILE_BYTES);
        tdm_load_2d(nb,                             Abase + k0 + GT_K, GT_K, 128, K, 3u, 3u);
        tdm_load_2d(nb + (unsigned)GEMM_TILE_BYTES, Wbase + k0 + GT_K, GT_K, 128, K, 3u, 3u);
        __builtin_amdgcn_s_wait_tensorcnt(2);   // current pair done, next in flight
      } else {
        __builtin_amdgcn_s_wait_tensorcnt(0);
      }
    }
    __syncthreads();
    lds_dirty(smem);   // TDM filled LDS invisibly: force real ds_loads below
    const bf16* pA = (const bf16*)(smem + cur * (unsigned)(2 * GEMM_TILE_BYTES));
    const bf16* pB = (const bf16*)(smem + cur * (unsigned)(2 * GEMM_TILE_BYTES)
                                        + (unsigned)GEMM_TILE_BYTES);
#else
  // Fallback: cooperative staging, single buffer.
  int sRow = tid >> 2;          // 0..63
  int sCol = (tid & 3) * 8;     // 0,8,16,24
  for (int k0 = 0; k0 < K; k0 += GT_K) {
    bf16* stA = (bf16*)smem;
    bf16* stB = (bf16*)(smem + GEMM_TILE_BYTES);
#pragma unroll
    for (int rr = 0; rr < 2; ++rr) {
      int r = sRow + rr * 64;
      *(v8bf*)&stA[r * GLD + sCol] = *(const v8bf*)&A[(rowBase + r) * K + k0 + sCol];
      *(v8bf*)&stB[r * GLD + sCol] = *(const v8bf*)&W[(colBase + r) * K + k0 + sCol];
    }
    if (k0 + GT_K < K) {   // global_prefetch_b8 of the next K tile
      __builtin_prefetch(&A[(rowBase + sRow) * K + k0 + GT_K + sCol], 0, 3);
      __builtin_prefetch(&W[(colBase + sRow) * K + k0 + GT_K + sCol], 0, 3);
    }
    __syncthreads();
    const bf16* pA = (const bf16*)smem;
    const bf16* pB = (const bf16*)(smem + GEMM_TILE_BYTES);
#endif
    FragBF16 a[2], b[4];
#pragma unroll
    for (int mf = 0; mf < 2; ++mf) {
      const bf16* p = &pA[(wm * 32 + mf * 16 + l16) * GLD + half * 8];
      a[mf].h[0] = *(const v8bf*)p;
      a[mf].h[1] = *(const v8bf*)(p + 16);
    }
#pragma unroll
    for (int nf = 0; nf < 4; ++nf) {
      const bf16* p = &pB[(wn * 64 + nf * 16 + l16) * GLD + half * 8];
      b[nf].h[0] = *(const v8bf*)p;
      b[nf].h[1] = *(const v8bf*)(p + 16);
    }
#pragma unroll
    for (int mf = 0; mf < 2; ++mf)
#pragma unroll
      for (int nf = 0; nf < 4; ++nf)
        acc[mf][nf] = wmma_bf16(a[mf], b[nf], acc[mf][nf]);
    __syncthreads();
  }

  // Epilogue: C row = base + r + 8*half, col = base + l16  (per ISA C/D layout)
#pragma unroll
  for (int mf = 0; mf < 2; ++mf)
#pragma unroll
    for (int nf = 0; nf < 4; ++nf) {
      long col = colBase + wn * 64 + nf * 16 + l16;
      float bv = bias ? bias[col] : 0.f;
#pragma unroll
      for (int r = 0; r < 8; ++r) {
        long row = rowBase + wm * 32 + mf * 16 + half * 8 + r;
        float vv = acc[mf][nf][r] + bv;
        if (relu) vv = fmaxf(vv, 0.f);
        long idx = row * N + col;
        if (resid) vv += resid[idx];
        if (outF) outF[idx] = vv;
        else      outB[idx] = (bf16)vv;
      }
    }
}

// ---------------------------------------------------------------------------
// Flash attention: one wave owns a 16-query tile; block = 8 waves shares
// K/V staging for one (b,h).  Online softmax in f32, score & PV via WMMA.
// K tiles staged by the TDM (wave0) while all threads build V^T in LDS.
// ---------------------------------------------------------------------------
#define FA_SK_OFF  0u        // 32 x 72 bf16 = 4608 B
#define FA_SVT_OFF 4608u     // 64 x 40 bf16 = 5120 B
#define FA_SP_OFF  9728u     // 8 waves x 16 x 40 bf16 = 10240 B

__global__ __launch_bounds__(256) void flash_attn(const bf16* __restrict__ q,
                                                  const bf16* __restrict__ k,
                                                  const bf16* __restrict__ v,
                                                  bf16* __restrict__ o) {
  __shared__ __align__(128) char smem[19968];

  int tid = threadIdx.x, lane = tid & 31, wave = tid >> 5;
  int half = lane >> 4, l16 = lane & 15;
  int bh = blockIdx.x >> 3, qblk = blockIdx.x & 7;
  int b = bh >> 4, h = bh & 15;
  long base = ((long)b * SS) * DD + h * DK;        // head slice, seq 0
  int qrow = qblk * 128 + wave * 16;

  bf16* sVT = (bf16*)(smem + FA_SVT_OFF);
  bf16* myP = (bf16*)(smem + FA_SP_OFF) + wave * 16 * 40;

  // Q fragments (A-matrix 16x32, two K-chunks covering dk=64)
  FragBF16 qf[2];
  {
    const bf16* qp = q + base + (long)(qrow + l16) * DD;
#pragma unroll
    for (int j = 0; j < 2; ++j) {
      qf[j].h[0] = *(const v8bf*)(qp + j * 32 + half * 8);
      qf[j].h[1] = *(const v8bf*)(qp + j * 32 + half * 8 + 16);
    }
  }

  float mstat[8], lstat[8];
  v8f oacc[4] = {};
#pragma unroll
  for (int r = 0; r < 8; ++r) { mstat[r] = -1e30f; lstat[r] = 0.f; }

  for (int kt = 0; kt < SS; kt += 32) {
    // Stage K tile (TDM: 32 keys x 64 dk, row stride D, pad 32+4 dwords)
#if HAS_TDM
    if (wave == 0)
      tdm_load_2d(FA_SK_OFF, k + base + (long)kt * DD, 64, 32, DD, 4u, 3u);
#endif
    // All threads stage V^T (and K too on the fallback path)
    {
      int key = tid >> 3, dc = (tid & 7) * 8;
#if !HAS_TDM
      bf16* sKw = (bf16*)(smem + FA_SK_OFF);
      const bf16* kp = k + base + (long)(kt + key) * DD + dc;
      *(v8bf*)&sKw[key * 72 + dc] = *(const v8bf*)kp;
#endif
      v8bf vv = *(const v8bf*)(v + base + (long)(kt + key) * DD + dc);
#pragma unroll
      for (int j = 0; j < 8; ++j) sVT[(dc + j) * 40 + key] = vv[j];
    }
#if HAS_TDM
    if (wave == 0) __builtin_amdgcn_s_wait_tensorcnt(0);
#endif
    __syncthreads();
    lds_dirty(smem);   // TDM wrote sK invisibly: force real ds_loads below
    const bf16* sK = (const bf16*)(smem + FA_SK_OFF);

    // Scores: S(16x32) = Q(16x64) * K^T, as two 16x16 WMMA tiles
    v8f sc[2];
#pragma unroll
    for (int nt = 0; nt < 2; ++nt) {
      FragBF16 kf0, kf1;
      const bf16* pk = &sK[(nt * 16 + l16) * 72 + half * 8];
      kf0.h[0] = *(const v8bf*)pk;        kf0.h[1] = *(const v8bf*)(pk + 16);
      kf1.h[0] = *(const v8bf*)(pk + 32); kf1.h[1] = *(const v8bf*)(pk + 48);
      v8f t = {};
      t = wmma_bf16(qf[0], kf0, t);
      t = wmma_bf16(qf[1], kf1, t);
      sc[nt] = t;
    }

    // Online softmax (rows split across the two 16-lane halves)
    float alpha[8];
#pragma unroll
    for (int r = 0; r < 8; ++r) {
      float s0 = sc[0][r] * 0.125f;   // 1/sqrt(64)
      float s1 = sc[1][r] * 0.125f;
      float tm = fmaxf(s0, s1);
#pragma unroll
      for (int m = 1; m < 16; m <<= 1) tm = fmaxf(tm, __shfl_xor(tm, m, 32));
      float mn = fmaxf(mstat[r], tm);
      float al = __expf(mstat[r] - mn);
      float p0 = __expf(s0 - mn), p1 = __expf(s1 - mn);
      float rs = p0 + p1;
#pragma unroll
      for (int m = 1; m < 16; m <<= 1) rs += __shfl_xor(rs, m, 32);
      lstat[r] = lstat[r] * al + rs;
      mstat[r] = mn;
      alpha[r] = al;
      myP[(r + 8 * half) * 40 + l16]      = (bf16)p0;
      myP[(r + 8 * half) * 40 + 16 + l16] = (bf16)p1;
    }

    // P (16x32) as an A fragment, via LDS round-trip (in-order DS per wave)
    FragBF16 pf;
    {
      const bf16* pp = &myP[l16 * 40 + half * 8];
      pf.h[0] = *(const v8bf*)pp;
      pf.h[1] = *(const v8bf*)(pp + 16);
    }

    // O += P * V   (4 WMMAs covering dk=64), with alpha rescale of C
#pragma unroll
    for (int df = 0; df < 4; ++df) {
#pragma unroll
      for (int r = 0; r < 8; ++r) oacc[df][r] *= alpha[r];
      FragBF16 vf;
      const bf16* pv = &sVT[(df * 16 + l16) * 40 + half * 8];
      vf.h[0] = *(const v8bf*)pv;
      vf.h[1] = *(const v8bf*)(pv + 16);
      oacc[df] = wmma_bf16(pf, vf, oacc[df]);
    }
    __syncthreads();
  }

  // Normalize and write O (bf16, [B,S,H,dk] == [BS,D])
#pragma unroll
  for (int r = 0; r < 8; ++r) {
    float inv = 1.f / lstat[r];
    long orow = base + (long)(qrow + r + 8 * half) * DD;
#pragma unroll
    for (int df = 0; df < 4; ++df)
      o[orow + df * 16 + l16] = (bf16)(oacc[df][r] * inv);
  }
}

// ---------------------------------------------------------------------------
// Host orchestration
// ---------------------------------------------------------------------------
extern "C" void kernel_launch(void* const* d_in, const int* in_sizes, int n_in,
                              void* d_out, int out_size, void* d_ws, size_t ws_size,
                              hipStream_t stream) {
  const int*   tokens = (const int*)  d_in[0];
  const float* emb    = (const float*)d_in[1];
  const float* attn_w = (const float*)d_in[2];
  const float* attn_b = (const float*)d_in[3];
  const float* ln_a   = (const float*)d_in[4];
  const float* ln_b   = (const float*)d_in[5];
  const float* w1     = (const float*)d_in[6];
  const float* b1     = (const float*)d_in[7];
  const float* w2     = (const float*)d_in[8];
  const float* b2     = (const float*)d_in[9];
  const float* fc_w   = (const float*)d_in[10];
  const float* fc_b   = (const float*)d_in[11];
  float* out = (float*)d_out;

  // Workspace carve-up (256B aligned regions)
  char* ws = (char*)d_ws;
  size_t off = 0;
  auto carve = [&](size_t bytes) -> char* {
    char* p = ws + off;
    off = (off + bytes + 255) & ~(size_t)255;
    return p;
  };
  float* x   = (float*)carve((size_t)BS * DD * 4);        // activations f32
  bf16*  y   = (bf16*) carve((size_t)BS * DD * 2);        // LN output bf16
  bf16*  qb  = (bf16*) carve((size_t)BS * DD * 2);
  bf16*  kb  = (bf16*) carve((size_t)BS * DD * 2);
  bf16*  vb  = (bf16*) carve((size_t)BS * DD * 2);
  bf16*  ob  = (bf16*) carve((size_t)BS * DD * 2);
  bf16*  hb  = (bf16*) carve((size_t)BS * FF * 2);        // FFN hidden
  bf16*  wAb = (bf16*) carve((size_t)LL * 4 * DD * DD * 2);
  bf16*  w1b = (bf16*) carve((size_t)LL * FF * DD * 2);
  bf16*  w2b = (bf16*) carve((size_t)LL * DD * FF * 2);
  bf16*  fcb = (bf16*) carve((size_t)VV * DD * 2);

  auto cvt = [&](const float* s, bf16* d, long n) {
    cvt_f32_to_bf16<<<dim3((unsigned)(n / (256 * 8))), dim3(256), 0, stream>>>(s, d, n);
  };
  // GEMM launcher: C = A * W^T (+bias)(+relu)(+resid)
  auto gemm = [&](const bf16* A, const bf16* W, const float* bias,
                  const float* resid, float* outF, bf16* outB,
                  int M, int N, int K, int relu) {
    gemm_bf16_wmma<<<dim3((unsigned)(N / 128), (unsigned)(M / 128)), dim3(256), 0, stream>>>(
        A, W, bias, resid, outF, outB, M, N, K, relu);
  };

  // One-time bf16 weight conversion
  cvt(attn_w, wAb, (long)LL * 4 * DD * DD);
  cvt(w1,     w1b, (long)LL * FF * DD);
  cvt(w2,     w2b, (long)LL * DD * FF);
  cvt(fc_w,   fcb, (long)VV * DD);

  // Embedding + positional encoding
  embed_pos<<<dim3(BS * DD / 4 / 256), dim3(256), 0, stream>>>(tokens, emb, x);

  for (int l = 0; l < LL; ++l) {
    const bf16* Wq = wAb + ((size_t)l * 4 + 0) * DD * DD;
    const bf16* Wk = wAb + ((size_t)l * 4 + 1) * DD * DD;
    const bf16* Wv = wAb + ((size_t)l * 4 + 2) * DD * DD;
    const bf16* Wo = wAb + ((size_t)l * 4 + 3) * DD * DD;
    const float* bq = attn_b + ((size_t)l * 4 + 0) * DD;
    const float* bk = attn_b + ((size_t)l * 4 + 1) * DD;
    const float* bv = attn_b + ((size_t)l * 4 + 2) * DD;
    const float* bo = attn_b + ((size_t)l * 4 + 3) * DD;

    // Attention sublayer
    layernorm_bf16<<<dim3(BS / 8), dim3(256), 0, stream>>>(
        x, ln_a + ((size_t)l * 2 + 0) * DD, ln_b + ((size_t)l * 2 + 0) * DD, y);
    gemm(y, Wq, bq, nullptr, nullptr, qb, BS, DD, DD, 0);
    gemm(y, Wk, bk, nullptr, nullptr, kb, BS, DD, DD, 0);
    gemm(y, Wv, bv, nullptr, nullptr, vb, BS, DD, DD, 0);
    flash_attn<<<dim3(BB * HH * (SS / 128)), dim3(256), 0, stream>>>(qb, kb, vb, ob);
    gemm(ob, Wo, bo, x, x, nullptr, BS, DD, DD, 0);   // residual into x

    // FFN sublayer
    layernorm_bf16<<<dim3(BS / 8), dim3(256), 0, stream>>>(
        x, ln_a + ((size_t)l * 2 + 1) * DD, ln_b + ((size_t)l * 2 + 1) * DD, y);
    gemm(y, w1b + (size_t)l * FF * DD, b1 + (size_t)l * FF,
         nullptr, nullptr, hb, BS, FF, DD, 1);        // +relu
    gemm(hb, w2b + (size_t)l * DD * FF, b2 + (size_t)l * DD,
         x, x, nullptr, BS, DD, FF, 0);               // residual into x
  }

  // Final vocab projection (no LN): logits = x * fc_w^T + fc_b
  cvt(x, y, (long)BS * DD);
  gemm(y, fcb, fc_b, nullptr, out, nullptr, BS, VV, DD, 0);
}
